// SwitchNetwork_68487548502016
// MI455X (gfx1250) — matrix operand order repaired and checked by reading, not compiled, not run on hardware
//
#include <hip/hip_runtime.h>

#define N_TOK 16384
#define DDIM  1024
#define NEXP  16
#define HDIM  4096
#define NCLS  1000
#define NCP   1024
#define CAP   2048   // 2*N/E

typedef __attribute__((ext_vector_type(16))) __bf16 v16bf;
typedef __attribute__((ext_vector_type(8)))  float  v8f;

static __device__ __forceinline__ unsigned short f2bf(float f) {
    unsigned int u = __float_as_uint(f);
    u = (u + 0x7FFFu + ((u >> 16) & 1u)) >> 16;   // round-to-nearest-even
    return (unsigned short)u;
}
static __device__ __forceinline__ float bf2f(unsigned short h) {
    return __uint_as_float(((unsigned int)h) << 16);
}
static __device__ __forceinline__ float gelu_tanh(float x) {
    const float c = 0.7978845608028654f;
    float t = tanhf(c * (x + 0.044715f * x * x * x));
    return 0.5f * x * (1.0f + t);
}

static __device__ __forceinline__ void wait_async0() {
#if __has_builtin(__builtin_amdgcn_s_wait_asynccnt)
    __builtin_amdgcn_s_wait_asynccnt(0);
#else
    asm volatile("s_wait_asynccnt 0x0" ::: "memory");
#endif
}

// Async memory->LDS copy of 16 bytes (per lane). The LDS destination operand
// is derived from the real LDS pointer (low 32 bits of the generic address =
// LDS byte offset, ISA 10.2), which also makes the LDS array escape into the
// asm so the compiler cannot fold subsequent ds_loads to undef.
static __device__ __forceinline__ void async_copy_b128(void* ldsPtr, const void* gaddr) {
    unsigned int lofs = (unsigned int)(unsigned long long)ldsPtr;
    unsigned long long ga = (unsigned long long)gaddr;
    asm volatile("global_load_async_to_lds_b128 %0, %1, off"
                 :: "v"(lofs), "v"(ga) : "memory");
}

// ---------------- gating: logits -> softmax -> argmax/gate -----------------
__global__ void gate_kernel(const float* __restrict__ x, const float* __restrict__ Wg,
                            float* __restrict__ probs, int* __restrict__ idx,
                            float* __restrict__ gate) {
    int wid = threadIdx.x >> 5, lane = threadIdx.x & 31;
    int n = blockIdx.x * 8 + wid;
    const float* xr = x + (size_t)n * DDIM;
    float acc[16];
    #pragma unroll
    for (int e = 0; e < 16; e++) acc[e] = 0.f;
    for (int k = lane; k < DDIM; k += 32) {
        float xv = xr[k];
        const float4* w4 = reinterpret_cast<const float4*>(Wg + (size_t)k * 16);
        #pragma unroll
        for (int q = 0; q < 4; q++) {
            float4 w = w4[q];
            acc[4*q+0] += xv * w.x; acc[4*q+1] += xv * w.y;
            acc[4*q+2] += xv * w.z; acc[4*q+3] += xv * w.w;
        }
    }
    #pragma unroll
    for (int e = 0; e < 16; e++)
        for (int off = 16; off > 0; off >>= 1)
            acc[e] += __shfl_xor(acc[e], off, 32);
    if (lane == 0) {
        float m = acc[0]; int am = 0;
        #pragma unroll
        for (int e = 1; e < 16; e++) if (acc[e] > m) { m = acc[e]; am = e; }
        float p[16]; float s = 0.f;
        #pragma unroll
        for (int e = 0; e < 16; e++) { p[e] = __expf(acc[e] - m); s += p[e]; }
        float inv = 1.f / s;
        #pragma unroll
        for (int e = 0; e < 16; e++) probs[(size_t)n * 16 + e] = p[e] * inv;
        idx[n]  = am;
        gate[n] = p[am] * inv;
    }
}

// ------------- capacity slot assignment: deterministic ballot scan ---------
__global__ void pos_kernel(const int* __restrict__ idx, int* __restrict__ pos) {
    int e = blockIdx.x;
    int lane = threadIdx.x;            // exactly one wave32
    int base = 0;
    for (int t0 = 0; t0 < N_TOK; t0 += 32) {
        int n = t0 + lane;
        bool pred = (idx[n] == e);
        unsigned int m32 = (unsigned int)__ballot(pred);
        int prefix = __popc(m32 & ((1u << lane) - 1u));
        if (pred) pos[n] = base + prefix;
        base += __popc(m32);
    }
}

// ------------------------- balancing loss (1 block) ------------------------
__global__ void loss_kernel(const float* __restrict__ probs, const int* __restrict__ idx,
                            float* __restrict__ out) {
    __shared__ float sred[256];
    int tid = threadIdx.x;
    float ps[16]; int cs[16];
    #pragma unroll
    for (int e = 0; e < 16; e++) { ps[e] = 0.f; cs[e] = 0; }
    for (int n = tid; n < N_TOK; n += 256) {
        cs[idx[n]]++;
        const float* pr = probs + (size_t)n * 16;
        #pragma unroll
        for (int q = 0; q < 16; q++) ps[q] += pr[q];
    }
    float dot = 0.f;
    for (int e = 0; e < 16; e++) {
        sred[tid] = ps[e]; __syncthreads();
        for (int s = 128; s > 0; s >>= 1) { if (tid < s) sred[tid] += sred[tid + s]; __syncthreads(); }
        float Pe = sred[0]; __syncthreads();
        sred[tid] = (float)cs[e]; __syncthreads();
        for (int s = 128; s > 0; s >>= 1) { if (tid < s) sred[tid] += sred[tid + s]; __syncthreads(); }
        float Ce = sred[0]; __syncthreads();
        dot += Pe * Ce;
    }
    if (tid == 0) out[0] = (float)NEXP * dot / ((float)N_TOK * (float)N_TOK);
}

// --------- transpose + fp32->bf16: in [E][K][N] f32 -> out [E][N][K] bf16 ---
// 32x32 LDS tile transpose; grid (N/32, K/32, E), 256 threads (32x8).
__global__ void transpose_to_bf16_kernel(const float* __restrict__ in,
                                         unsigned short* __restrict__ out,
                                         int K, int N) {
    __shared__ float tile[32][33];
    int e = blockIdx.z;
    const float* src = in + (size_t)e * K * N;
    unsigned short* dst = out + (size_t)e * K * N;
    int n0 = blockIdx.x * 32, k0 = blockIdx.y * 32;
    int tx = threadIdx.x & 31, ty = threadIdx.x >> 5;
    #pragma unroll
    for (int r = 0; r < 4; r++) {
        int k = ty + r * 8;
        tile[k][tx] = src[(size_t)(k0 + k) * N + n0 + tx];
    }
    __syncthreads();
    #pragma unroll
    for (int r = 0; r < 4; r++) {
        int n = ty + r * 8;
        dst[(size_t)(n0 + n) * K + k0 + tx] = f2bf(tile[tx][n]);
    }
}

// Wl [D][NCLS] f32 -> wlBt [NCP][D] bf16 (transposed, rows >= NCLS zero-pad)
__global__ void wl_t_pad_kernel(const float* __restrict__ Wl, unsigned short* __restrict__ wlBt) {
    long long i = blockIdx.x * (long long)blockDim.x + threadIdx.x;
    long long stride = (long long)gridDim.x * blockDim.x;
    long long total = (long long)NCP * DDIM;
    for (; i < total; i += stride) {
        int n = (int)(i >> 10), k = (int)(i & (DDIM - 1));
        wlBt[i] = (n < NCLS) ? f2bf(Wl[(size_t)k * NCLS + n]) : (unsigned short)0;
    }
}

// -------------------- dispatch kept tokens into [E,C,D] --------------------
__global__ void dispatch_kernel(const float* __restrict__ x, const int* __restrict__ idx,
                                const int* __restrict__ pos, unsigned short* __restrict__ dispB) {
    int n = blockIdx.x;
    int p = pos[n];
    if (p >= CAP) return;                 // dropped (over capacity)
    int e = idx[n];
    const float* src = x + (size_t)n * DDIM;
    unsigned short* dst = dispB + ((size_t)e * CAP + p) * DDIM;
    for (int c = threadIdx.x; c < DDIM; c += blockDim.x) dst[c] = f2bf(src[c]);
}

// ---------------- combine: gather expert output, scale by gate -------------
__global__ void combine_kernel(const unsigned short* __restrict__ outeB, const int* __restrict__ idx,
                               const int* __restrict__ pos, const float* __restrict__ gate,
                               unsigned short* __restrict__ yB) {
    int n = blockIdx.x;
    int p = pos[n], e = idx[n];
    float gv = (p < CAP) ? gate[n] : 0.f;
    int pc = (p < CAP) ? p : (CAP - 1);
    const unsigned short* src = outeB + ((size_t)e * CAP + pc) * DDIM;
    unsigned short* dst = yB + (size_t)n * DDIM;
    for (int c = threadIdx.x; c < DDIM; c += blockDim.x) dst[c] = f2bf(bf2f(src[c]) * gv);
}

// ------------------- generic tiled bf16 WMMA GEMM (128x128x32) -------------
// A [M][K] bf16 row-major; B passed PRE-TRANSPOSED [N][K] bf16 so both tiles
// are 128 rows x 32 contiguous bf16. Both tiles stream memory->LDS with
// global_load_async_to_lds_b128 (ASYNCcnt), double-buffered so tile k+1
// copies overlap WMMA compute of tile k. 8 waves, 2x4 WMMA subtiles each.
__global__ __launch_bounds__(256) void gemm_bf16(
    const unsigned short* __restrict__ A, const unsigned short* __restrict__ B,
    const float* __restrict__ bias, void* __restrict__ Cout,
    int M, int Ncol, int K,
    long long strideA, long long strideB, long long strideC, long long strideBias,
    int ldA, int ldB, int ldC, int ncols, int mode) {
    (void)M; (void)Ncol;
    __shared__ __align__(16) unsigned short As[2][128 * 40];
    __shared__ __align__(16) unsigned short Bs[2][128 * 40];
    int e = blockIdx.z;
    A += (size_t)e * strideA;
    B += (size_t)e * strideB;
    const float* bptr = bias + (size_t)e * strideBias;
    float* Cf = (float*)Cout + (size_t)e * strideC;
    unsigned short* Ch = (unsigned short*)Cout + (size_t)e * strideC;

    int nBase = blockIdx.x * 128, mBase = blockIdx.y * 128;
    int tid = threadIdx.x;
    int w = tid >> 5, lid = tid & 31;
    int waveM = w & 3, waveN = w >> 2;     // 4 x 2 waves
    int g = lid >> 4, ln = lid & 15;

    v8f acc[2][4] = {};
    int ksteps = K >> 5;

    // issue async copies for one K-tile into buffer `buf`
    auto issue = [&](int k0, int buf) {
        #pragma unroll
        for (int h = 0; h < 2; h++) {
            int c = tid + h * 256;
            int row = c >> 2, seg = c & 3;
            async_copy_b128(&As[buf][row * 40 + seg * 8],
                            &A[(size_t)(mBase + row) * ldA + k0 + seg * 8]);
            async_copy_b128(&Bs[buf][row * 40 + seg * 8],
                            &B[(size_t)(nBase + row) * ldB + k0 + seg * 8]);
        }
    };

    issue(0, 0);
    wait_async0();
    __syncthreads();

    for (int kk = 0; kk < ksteps; kk++) {
        int buf = kk & 1;
        if (kk + 1 < ksteps) issue((kk + 1) << 5, buf ^ 1);
        if (kk + 2 < ksteps) {  // global_prefetch_b8 for the tile after next
            int kp = (kk + 2) << 5;
            __builtin_prefetch(&A[(size_t)(mBase + (tid & 127)) * ldA + kp], 0, 0);
            __builtin_prefetch(&B[(size_t)(nBase + (tid & 127)) * ldB + kp], 0, 0);
        }

        union FA { uint4 q[2]; v16bf v; };
        FA af[2], bfrag[4];
        #pragma unroll
        for (int i = 0; i < 2; i++) {
            // A frag: lane holds row m = ln; halves [0..7]=K(g*8..), [8..15]=K(16+g*8..)
            int moff = (waveM * 32 + i * 16 + ln) * 40 + g * 8;
            af[i].q[0] = *reinterpret_cast<const uint4*>(&As[buf][moff]);
            af[i].q[1] = *reinterpret_cast<const uint4*>(&As[buf][moff + 16]);
        }
        #pragma unroll
        for (int j = 0; j < 4; j++) {
            // B frag: lane holds col n = ln; halves [0..7]=K(g*8..), [8..15]=K(16+g*8..)
            int noff = (waveN * 64 + j * 16 + ln) * 40 + g * 8;
            bfrag[j].q[0] = *reinterpret_cast<const uint4*>(&Bs[buf][noff]);
            bfrag[j].q[1] = *reinterpret_cast<const uint4*>(&Bs[buf][noff + 16]);
        }
        #pragma unroll
        for (int i = 0; i < 2; i++)
            #pragma unroll
            for (int j = 0; j < 4; j++)
                acc[i][j] = __builtin_amdgcn_wmma_f32_16x16x32_bf16(
                    false, af[i].v, false, bfrag[j].v, (short)0, acc[i][j], false, false);

        wait_async0();      // next tile landed (overlapped with WMMA above)
        __syncthreads();    // everyone done reading buf before it is refilled
    }

    // epilogue: C/D layout — VGPR r: M = r + 8*(lane/16); N = lane%16
    #pragma unroll
    for (int i = 0; i < 2; i++) {
        int rowb = mBase + waveM * 32 + i * 16 + 8 * g;
        #pragma unroll
        for (int j = 0; j < 4; j++) {
            int col = nBase + waveN * 64 + j * 16 + ln;
            if (col >= ncols) continue;
            float bv = bptr[col];
            #pragma unroll
            for (int r = 0; r < 8; r++) {
                float v = acc[i][j][r] + bv;
                size_t off = (size_t)(rowb + r) * ldC + col;
                if (mode == 0)       Cf[off] = v;
                else if (mode == 1)  Ch[off] = f2bf(v);
                else                 Ch[off] = f2bf(gelu_tanh(v));
            }
        }
    }
}

// ---------------------------------------------------------------------------
extern "C" void kernel_launch(void* const* d_in, const int* in_sizes, int n_in,
                              void* d_out, int out_size, void* d_ws, size_t ws_size,
                              hipStream_t stream) {
    (void)in_sizes; (void)n_in; (void)out_size; (void)ws_size;
    const float* x  = (const float*)d_in[0];
    const float* Wg = (const float*)d_in[1];
    const float* W1 = (const float*)d_in[2];
    const float* b1 = (const float*)d_in[3];
    const float* W2 = (const float*)d_in[4];
    const float* b2 = (const float*)d_in[5];
    const float* Wl = (const float*)d_in[6];
    const float* bl = (const float*)d_in[7];
    float* out = (float*)d_out;

    char* ws = (char*)d_ws;
    size_t off = 0;
    auto alloc = [&](size_t bytes) { void* p = ws + off; off += (bytes + 255) & ~(size_t)255; return p; };
    float*          probs = (float*)alloc((size_t)N_TOK * 16 * 4);
    int*            idx   = (int*)alloc((size_t)N_TOK * 4);
    int*            pos   = (int*)alloc((size_t)N_TOK * 4);
    float*          gate  = (float*)alloc((size_t)N_TOK * 4);
    unsigned short* dispB = (unsigned short*)alloc((size_t)NEXP * CAP * DDIM * 2);
    unsigned short* w1Bt  = (unsigned short*)alloc((size_t)NEXP * DDIM * HDIM * 2);  // [E][H][D]
    unsigned short* w2Bt  = (unsigned short*)alloc((size_t)NEXP * HDIM * DDIM * 2);  // [E][D][H]
    unsigned short* wlBt  = (unsigned short*)alloc((size_t)NCP * DDIM * 2);          // [NCP][D]
    unsigned short* hB    = (unsigned short*)alloc((size_t)NEXP * CAP * HDIM * 2);
    unsigned short* outeB = (unsigned short*)alloc((size_t)NEXP * CAP * DDIM * 2);
    unsigned short* yB    = (unsigned short*)alloc((size_t)N_TOK * DDIM * 2);

    gate_kernel<<<dim3(N_TOK / 8), dim3(256), 0, stream>>>(x, Wg, probs, idx, gate);
    pos_kernel<<<dim3(NEXP), dim3(32), 0, stream>>>(idx, pos);
    loss_kernel<<<dim3(1), dim3(256), 0, stream>>>(probs, idx, out + (size_t)N_TOK * NCLS);

    // pre-transpose weights to [N][K] bf16 (done once; K then contiguous for B tiles)
    transpose_to_bf16_kernel<<<dim3(HDIM / 32, DDIM / 32, NEXP), dim3(256), 0, stream>>>(
        W1, w1Bt, DDIM, HDIM);
    transpose_to_bf16_kernel<<<dim3(DDIM / 32, HDIM / 32, NEXP), dim3(256), 0, stream>>>(
        W2, w2Bt, HDIM, DDIM);
    wl_t_pad_kernel<<<dim3(1024), dim3(256), 0, stream>>>(Wl, wlBt);
    dispatch_kernel<<<dim3(N_TOK), dim3(128), 0, stream>>>(x, idx, pos, dispB);

    // GEMM1: h = gelu(disp @ W1 + b1)   [E] 2048x4096x1024
    gemm_bf16<<<dim3(HDIM / 128, CAP / 128, NEXP), dim3(256), 0, stream>>>(
        dispB, w1Bt, b1, (void*)hB, CAP, HDIM, DDIM,
        (long long)CAP * DDIM, (long long)DDIM * HDIM, (long long)CAP * HDIM, (long long)HDIM,
        DDIM, DDIM, HDIM, HDIM, 2);
    // GEMM2: out_e = h @ W2 + b2        [E] 2048x1024x4096
    gemm_bf16<<<dim3(DDIM / 128, CAP / 128, NEXP), dim3(256), 0, stream>>>(
        hB, w2Bt, b2, (void*)outeB, CAP, DDIM, HDIM,
        (long long)CAP * HDIM, (long long)HDIM * DDIM, (long long)CAP * DDIM, (long long)DDIM,
        HDIM, HDIM, DDIM, DDIM, 1);

    combine_kernel<<<dim3(N_TOK), dim3(128), 0, stream>>>(outeB, idx, pos, gate, yB);

    // GEMM3: logits = y @ Wl + bl       16384x1024(1000)x1024, fp32 out
    gemm_bf16<<<dim3(NCP / 128, N_TOK / 128, 1), dim3(256), 0, stream>>>(
        yB, wlBt, bl, d_out, N_TOK, NCP, DDIM,
        0LL, 0LL, 0LL, 0LL,
        DDIM, DDIM, NCLS, NCLS, 0);
}
